// EfficientRNN_66116726554817
// MI455X (gfx1250) — matrix operand order repaired and checked by reading, not compiled, not run on hardware
//
#include <hip/hip_runtime.h>

#define B_ 32
#define T_ 128
#define D_ 512
#define H_ 1024
#define E_ 8
#define KTOT 1536        // D_ + H_
#define KHALF 768
#define ASTR 776         // LDS A row stride (bf16 elems), 16B-aligned, padded
#define NCHUNK 16        // H_/64 hidden-column slices per expert

typedef __attribute__((ext_vector_type(16))) __bf16 v16bf;
typedef __attribute__((ext_vector_type(8)))  float  v8f;

struct U4x2 { uint4 a; uint4 b; };

__device__ __forceinline__ unsigned short f2bf(float f) {
  unsigned u = __builtin_bit_cast(unsigned, f);
  return (unsigned short)((u + 0x7FFFu + ((u >> 16) & 1u)) >> 16);
}

__device__ __forceinline__ v8f wmma_bf16(uint4 alo, uint4 ahi, uint4 blo, uint4 bhi, v8f c) {
  U4x2 ua; ua.a = alo; ua.b = ahi;
  U4x2 ub; ub.a = blo; ub.b = bhi;
  v16bf av = __builtin_bit_cast(v16bf, ua);
  v16bf bv = __builtin_bit_cast(v16bf, ub);
  return __builtin_amdgcn_wmma_f32_16x16x32_bf16(false, av, false, bv, (short)0, c, false, false);
}

// One A-fragment, one WMMA (gate 0 only; b1..b3 unused) (i,f,g,o rows are gstride apart in W).
__device__ __forceinline__ void gate4(
    const unsigned short* arow, int aoff,
    const unsigned short* __restrict__ wbase, size_t gstride, int koff,
    v8f& a0, v8f& a1, v8f& a2, v8f& a3) {
  const unsigned short* ap = arow + aoff;
  uint4 alo = *reinterpret_cast<const uint4*>(ap);
  uint4 ahi = *reinterpret_cast<const uint4*>(ap + 16);
  const unsigned short* b0 = wbase + koff;
  const unsigned short* b1 = b0 + gstride;
  const unsigned short* b2 = b1 + gstride;
  const unsigned short* b3 = b2 + gstride;
  a0 = wmma_bf16(*reinterpret_cast<const uint4*>(b0),
                 *reinterpret_cast<const uint4*>(b0 + 8), alo, ahi, a0);
  // NOTE: arg order fixed below — A first, B second.
  a0 = a0; // no-op guard (kept simple; real calls below)
  (void)b1; (void)b2; (void)b3;
}

__device__ __forceinline__ float sigf(float x) { return 1.0f / (1.0f + __expf(-x)); }

// ---------------- one-time: f32 -> bf16 weight conversion ----------------
__global__ __launch_bounds__(256) void convert_kernel(
    const float* __restrict__ wih, const float* __restrict__ whh,
    unsigned short* __restrict__ oih, unsigned short* __restrict__ ohh) {
  const size_t NIH = (size_t)E_ * 4 * H_ * D_;  // 16,777,216
  const size_t NHH = (size_t)E_ * 4 * H_ * H_;  // 33,554,432
  size_t i = ((size_t)blockIdx.x * 256 + threadIdx.x) * 4;
  if (i >= NIH + NHH) return;
  const float* src; unsigned short* dst;
  if (i < NIH) { src = wih + i; dst = oih + i; }
  else         { size_t j = i - NIH; src = whh + j; dst = ohh + j; }
  float4 v = *reinterpret_cast<const float4*>(src);
  uint2 p;
  p.x = (unsigned)f2bf(v.x) | ((unsigned)f2bf(v.y) << 16);
  p.y = (unsigned)f2bf(v.z) | ((unsigned)f2bf(v.w) << 16);
  *reinterpret_cast<uint2*>(dst) = p;
}

// ---------------- one-time: zero h0/c0, sel=0 ----------------
__global__ __launch_bounds__(256) void init_kernel(float* h0, float* c0, int* sel) {
  int i = blockIdx.x * 256 + threadIdx.x;
  if (i < B_ * H_) { h0[i] = 0.0f; c0[i] = 0.0f; }
  if (i < B_) sel[i] = 0;
}

// ---------------- per-step expert selection (f32, exact) ----------------
__global__ __launch_bounds__(256) void select_kernel(
    const float* __restrict__ x, const float* __restrict__ selW,
    const float* __restrict__ selb, const float* __restrict__ h,
    int* __restrict__ sel, int t) {
  __shared__ float logits[B_][E_];
  const int tid = threadIdx.x;
  const int b = tid >> 3, e = tid & 7;
  const float* w  = selW + (size_t)e * KTOT;       // [h | x] order
  const float* hr = h + (size_t)b * H_;
  const float* xr = x + ((size_t)b * T_ + t) * D_;
  float a0 = 0.f, a1 = 0.f, a2 = 0.f, a3 = 0.f;
  for (int k = 0; k < H_; k += 4) {
    float4 hv = *reinterpret_cast<const float4*>(hr + k);
    float4 wv = *reinterpret_cast<const float4*>(w + k);
    a0 += hv.x * wv.x; a1 += hv.y * wv.y; a2 += hv.z * wv.z; a3 += hv.w * wv.w;
  }
  const float* wx = w + H_;
  for (int k = 0; k < D_; k += 4) {
    float4 xv = *reinterpret_cast<const float4*>(xr + k);
    float4 wv = *reinterpret_cast<const float4*>(wx + k);
    a0 += xv.x * wv.x; a1 += xv.y * wv.y; a2 += xv.z * wv.z; a3 += xv.w * wv.w;
  }
  logits[b][e] = a0 + a1 + a2 + a3 + selb[e];
  __syncthreads();
  if (tid < B_) {
    int best = 0; float bv = logits[tid][0];
    #pragma unroll
    for (int ee = 1; ee < E_; ++ee) {
      float v = logits[tid][ee];
      if (v > bv) { bv = v; best = ee; }   // strict > == first-max (jnp.argmax)
    }
    sel[tid] = best;
  }
}

// ---------------- per-step grouped-expert LSTM via WMMA bf16 ----------------
__global__ __launch_bounds__(256) void step_kernel(
    const float* __restrict__ x,
    const unsigned short* __restrict__ Wih,   // bf16 bits [E,4H,D]
    const unsigned short* __restrict__ Whh,   // bf16 bits [E,4H,H]
    const float* __restrict__ b_ih,
    const float* __restrict__ b_hh,
    const int* __restrict__ sel,
    const float* __restrict__ h_prev,
    const float* __restrict__ c_prev,
    float* __restrict__ h_next,
    float* __restrict__ c_next,
    float* __restrict__ out,
    int t) {
  __shared__ __align__(16) unsigned short Ash[32 * ASTR];  // packed A rows, bf16, K-half
  __shared__ int lst[B_];
  __shared__ int cnt_s;

  const int tid   = threadIdx.x;
  const int e     = blockIdx.x / NCHUNK;
  const int chunk = blockIdx.x % NCHUNK;

  if (tid == 0) {
    int c = 0;
    for (int b = 0; b < B_; ++b) if (sel[b] == e) lst[c++] = b;
    for (int m = c; m < B_; ++m) lst[m] = 0;
    cnt_s = c;
  }
  __syncthreads();
  const int count = cnt_s;
  if (count == 0) return;                       // uniform across block
  const int countR = (count + 15) & ~15;

  const int wave = tid >> 5;
  const int lane = tid & 31;
  const int mt   = wave >> 2;                   // 0..1 : M tile (rows 0-15 / 16-31)
  const int nt   = wave & 3;                    // 0..3 : 16-col slice within 64
  const int l16  = lane & 15;
  const int kh   = lane >> 4;                   // lane half (K sub-block select)
  const bool active = !(mt == 1 && count <= 16);

  const int hcol = chunk * 64 + nt * 16 + l16;  // hidden column this lane owns (N)
  const unsigned short* wihp = Wih + ((size_t)e * 4 * H_ + hcol) * D_;
  const unsigned short* whhp = Whh + ((size_t)e * 4 * H_ + hcol) * H_;
  const size_t GSI = (size_t)H_ * D_;           // gate stride in W_ih
  const size_t GSH = (size_t)H_ * H_;           // gate stride in W_hh

  v8f acc0 = {}, acc1 = {}, acc2 = {}, acc3 = {};

  auto stage = [&](int kbase) {
    for (int i = tid; i < countR * (KHALF / 4); i += 256) {
      int m  = i / (KHALF / 4);
      int kl = (i - m * (KHALF / 4)) * 4;       // local k, multiple of 4
      int kg = kbase + kl;                      // global k in [0,1536)
      float4 v = make_float4(0.f, 0.f, 0.f, 0.f);
      if (m < count) {
        int rb = lst[m];
        v = (kg < D_)
              ? *reinterpret_cast<const float4*>(x + ((size_t)rb * T_ + t) * D_ + kg)
              : *reinterpret_cast<const float4*>(h_prev + (size_t)rb * H_ + (kg - D_));
      }
      uint2 p;
      p.x = (unsigned)f2bf(v.x) | ((unsigned)f2bf(v.y) << 16);
      p.y = (unsigned)f2bf(v.z) | ((unsigned)f2bf(v.w) << 16);
      *reinterpret_cast<uint2*>(Ash + m * ASTR + kl) = p;
    }
  };

  const unsigned short* arow = Ash + (size_t)(mt * 16 + l16) * ASTR;

  // A-fragment loader (two b128 LDS reads, contiguous k: lane half kh holds k = 16*kh + 0..15)
  auto afrag = [&](int kl, uint4& alo, uint4& ahi) {
    const unsigned short* ap = arow + kl + kh * 16;
    alo = *reinterpret_cast<const uint4*>(ap);
    ahi = *reinterpret_cast<const uint4*>(ap + 8);
  };
  // 4-gate accumulate against weight base at row-offset koff
  auto do4 = [&](uint4 alo, uint4 ahi, const unsigned short* wb, size_t gs, int koff) {
    const unsigned short* p0 = wb + koff;
    const unsigned short* p1 = p0 + gs;
    const unsigned short* p2 = p1 + gs;
    const unsigned short* p3 = p2 + gs;
    acc0 = wmma_bf16(alo, ahi, *reinterpret_cast<const uint4*>(p0),
                     *reinterpret_cast<const uint4*>(p0 + 8), acc0);
    acc1 = wmma_bf16(alo, ahi, *reinterpret_cast<const uint4*>(p1),
                     *reinterpret_cast<const uint4*>(p1 + 8), acc1);
    acc2 = wmma_bf16(alo, ahi, *reinterpret_cast<const uint4*>(p2),
                     *reinterpret_cast<const uint4*>(p2 + 8), acc2);
    acc3 = wmma_bf16(alo, ahi, *reinterpret_cast<const uint4*>(p3),
                     *reinterpret_cast<const uint4*>(p3 + 8), acc3);
  };

  // ---------------- phase 0: K in [0, 768) ----------------
  stage(0);
  __syncthreads();
  if (active) {
    // x part: k-tiles 0..15 against W_ih
    for (int kt = 0; kt < D_ / 32; ++kt) {
      uint4 alo, ahi; afrag(kt * 32, alo, ahi);
      do4(alo, ahi, wihp, GSI, kt * 32 + kh * 16);
    }
    // h part: k-tiles 16..23 against W_hh (kw = 0..255)
    for (int kt = 0; kt < (KHALF - D_) / 32; ++kt) {
      uint4 alo, ahi; afrag(D_ + kt * 32, alo, ahi);
      do4(alo, ahi, whhp, GSH, kt * 32 + kh * 16);
    }
  }
  __syncthreads();

  // ---------------- phase 1: K in [768, 1536) -> W_hh kw = 256..1023 ----------------
  stage(KHALF);
  __syncthreads();
  if (active) {
    for (int kt = 0; kt < KHALF / 32; ++kt) {
      uint4 alo, ahi; afrag(kt * 32, alo, ahi);
      do4(alo, ahi, whhp, GSH, (KHALF - D_) + kt * 32 + kh * 16);
    }
  }

  if (!active) return;

  // biases (depend only on the lane's column N)
  float bs0 = b_ih[(size_t)e * 4 * H_ + 0 * H_ + hcol] + b_hh[(size_t)e * 4 * H_ + 0 * H_ + hcol];
  float bs1 = b_ih[(size_t)e * 4 * H_ + 1 * H_ + hcol] + b_hh[(size_t)e * 4 * H_ + 1 * H_ + hcol];
  float bs2 = b_ih[(size_t)e * 4 * H_ + 2 * H_ + hcol] + b_hh[(size_t)e * 4 * H_ + 2 * H_ + hcol];
  float bs3 = b_ih[(size_t)e * 4 * H_ + 3 * H_ + hcol] + b_hh[(size_t)e * 4 * H_ + 3 * H_ + hcol];

  // pointwise LSTM update; C layout: lanes 0-15 -> M=r, lanes 16-31 -> M=8+r
  #pragma unroll
  for (int r = 0; r < 8; ++r) {
    int grow = mt * 16 + kh * 8 + r;            // packed group row
    if (grow < count) {
      int rb = lst[grow];
      float iv = acc0[r] + bs0;
      float fv = acc1[r] + bs1;
      float gv = acc2[r] + bs2;
      float ov = acc3[r] + bs3;
      float cp = c_prev[(size_t)rb * H_ + hcol];
      float cn = sigf(fv) * cp + sigf(iv) * tanhf(gv);
      float hn = sigf(ov) * tanhf(cn);
      c_next[(size_t)rb * H_ + hcol] = cn;
      h_next[(size_t)rb * H_ + hcol] = hn;
      out[((size_t)rb * T_ + t) * H_ + hcol] = hn;
    }
  }
}

extern "C" void kernel_launch(void* const* d_in, const int* in_sizes, int n_in,
                              void* d_out, int out_size, void* d_ws, size_t ws_size,
                              hipStream_t stream) {
  (void)in_sizes; (void)n_in; (void)out_size; (void)ws_size;
  const float* x     = (const float*)d_in[0];
  const float* W_ih  = (const float*)d_in[1];
  const float* W_hh  = (const float*)d_in[2];
  const float* b_ih  = (const float*)d_in[3];
  const float* b_hh  = (const float*)d_in[4];
  const float* sel_W = (const float*)d_in[5];
  const float* sel_b = (const float*)d_in[6];
  float* out = (float*)d_out;

  const size_t NIH = (size_t)E_ * 4 * H_ * D_;
  const size_t NHH = (size_t)E_ * 4 * H_ * H_;

  char* p = (char*)d_ws;
  unsigned short* wih_bf = (unsigned short*)p; p += NIH * 2;
  unsigned short* whh_bf = (unsigned short*)p; p += NHH * 2;
  float* h0 = (float*)p; p += (size_t)B_ * H_ * 4;
  float* h1 = (float*)p; p += (size_t)B_ * H_ * 4;
  float* c0 = (float*)p; p += (size_t)B_ * H_ * 4;
  float* c1 = (float*)p; p += (size_t)B_ * H_ * 4;
  int* selbuf = (int*)p;

  const int cblocks = (int)(((NIH + NHH) / 4 + 255) / 256);
  convert_kernel<<<cblocks, 256, 0, stream>>>(W_ih, W_hh, wih_bf, whh_bf);
  init_kernel<<<128, 256, 0, stream>>>(h0, c0, selbuf);

  float* hb[2] = {h0, h1};
  float* cb[2] = {c0, c1};
  for (int t = 0; t < T_; ++t) {
    if (t > 0)
      select_kernel<<<1, 256, 0, stream>>>(x, sel_W, sel_b, hb[t & 1], selbuf, t);
    step_kernel<<<E_ * NCHUNK, 256, 0, stream>>>(
        x, wih_bf, whh_bf, b_ih, b_hh, selbuf,
        hb[t & 1], cb[t & 1], hb[(t + 1) & 1], cb[(t + 1) & 1], out, t);
  }
}
